// SWABlock_62792421867581
// MI455X (gfx1250) — matrix operand order repaired and checked by reading, not compiled, not run on hardware
//
#include <hip/hip_runtime.h>
#include <stdint.h>

#define NTOK 32
#define NH   16
#define CDIM 512
#define HD   32
#define NWIN 4096
#define CPB_HID 512
#define BN   (NWIN * NTOK)   // 131072 rows

typedef __attribute__((ext_vector_type(16))) __bf16 v16bf;
typedef __attribute__((ext_vector_type(8)))  float  v8f;

#if defined(__has_builtin)
#if __has_builtin(__builtin_amdgcn_global_load_async_to_lds_b32)
#define HAVE_ASYNC_LDS 1
typedef __attribute__((address_space(1))) int gint_as1;
typedef __attribute__((address_space(3))) int sint_as3;
#endif
#endif

union FragAB {
    v16bf v;
    uint4 q[2];   // 2 x 16B chunks
};

__device__ __forceinline__ uint16_t f2bf(float f) {
    uint32_t u = __float_as_uint(f);
    u += 0x7FFFu + ((u >> 16) & 1u);          // round-to-nearest-even
    return (uint16_t)(u >> 16);
}

__device__ __forceinline__ v8f wmma_bf16(const FragAB& a, const FragAB& b, v8f c) {
    return __builtin_amdgcn_wmma_f32_16x16x32_bf16(
        /*neg_a=*/false, a.v, /*neg_b=*/false, b.v,
        /*c_mod=*/(short)0, c, /*reuse_a=*/false, /*reuse_b=*/false);
}

// ---------------------------------------------------------------------------
// K0: convert qkv_w [1536][512] and out_w [512][512] f32 -> bf16 (row-major)
// ---------------------------------------------------------------------------
__global__ __launch_bounds__(256) void prep_convert(
    const float* __restrict__ qkv_w, const float* __restrict__ out_w,
    uint16_t* __restrict__ qkvw_bf, uint16_t* __restrict__ outw_bf)
{
    int i = blockIdx.x * 256 + threadIdx.x;
    if (i < 1536 * 512) qkvw_bf[i] = f2bf(qkv_w[i]);
    int j = i - 1536 * 512;
    if (j >= 0 && j < 512 * 512) outw_bf[j] = f2bf(out_w[j]);
}

// ---------------------------------------------------------------------------
// K0b: bias table (only 63 rows of the CPB table are reachable) + head scales
// biasTab[d*16+h] = 16*sigmoid( relu(tbl_d @ w1^T + b1) @ w2[h] ),  d = i-j+31
// ---------------------------------------------------------------------------
__global__ __launch_bounds__(256) void prep_bias(
    const float* __restrict__ w1, const float* __restrict__ b1,
    const float* __restrict__ w2, const float* __restrict__ logit_scale,
    float* __restrict__ biasTab, float* __restrict__ scaleTab)
{
    int gid = blockIdx.x * 256 + threadIdx.x;
    if (gid < 63 * 16) {
        int d = gid >> 4, h = gid & 15;
        // flattened table row k (k<63) corresponds to coords (-31,-31,k-31)
        float s0  = -8.0f;
        float c01 = copysignf(log2f(fabsf(s0) + 1.0f) * (1.0f / 3.0f), s0);
        float s2  = (float)(d - 31) * (8.0f / 31.0f);
        float c2  = copysignf(log2f(fabsf(s2) + 1.0f) * (1.0f / 3.0f), s2);
        float acc = 0.0f;
        for (int k = 0; k < CPB_HID; ++k) {
            float hk = c01 * w1[k * 3 + 0] + c01 * w1[k * 3 + 1]
                     + c2  * w1[k * 3 + 2] + b1[k];
            hk = fmaxf(hk, 0.0f);
            acc += hk * w2[h * CPB_HID + k];
        }
        biasTab[gid] = 16.0f / (1.0f + __expf(-acc));
    } else if (gid < 63 * 16 + NH) {
        int h = gid - 63 * 16;
        scaleTab[h] = __expf(fminf(logit_scale[h], 4.60517019f)); // ln(100)
    }
}

// ---------------------------------------------------------------------------
// K1: fused QKV + cosine attention. 1 block = 2 windows, 8 waves.
// Each weight byte is streamed from L2 exactly once per block (B fragments
// reused across all 4 m-tiles of the two windows).
// ---------------------------------------------------------------------------
__global__ __launch_bounds__(256) void attn_kernel(
    const float*    __restrict__ x,        // [4096][32][512]
    const uint16_t* __restrict__ qkvw_bf,  // [1536][512] bf16 row-major
    const float*    __restrict__ biasTab,  // [63][16]
    const float*    __restrict__ scaleTab, // [16]
    uint16_t*       __restrict__ ao)       // [131072][512] bf16
{
    // padded strides keep 16B alignment and rotate LDS banks between rows
    __shared__ uint16_t xs[64 * 520];     // 2 windows of x, bf16, stride 1040B=65*16B
    __shared__ float    biasL[63 * 16];
    __shared__ float    qf[64 * 36];      // raw q head f32 (both windows)
    __shared__ float    kf[64 * 36];      // raw k head f32
    __shared__ uint16_t qh[64 * 40];      // normalized*scale q bf16 [m][d]
    __shared__ uint16_t kh[64 * 40];      // normalized k bf16     [n][d]
    __shared__ uint16_t vh[64 * 40];      // v transposed bf16, per window [d][m]
    __shared__ uint16_t sp[64 * 40];      // softmax probs bf16    [m][n]
    __shared__ float    sf[64 * 36];      // attention logits f32

    const int tid  = threadIdx.x;
    const int lane = tid & 31;
    const int wave = tid >> 5;
    const int hs   = lane >> 4;
    const int l15  = lane & 15;
    const int wb   = blockIdx.x * 2;      // first window of the pair

    // ---- bias table -> LDS (async direct-to-LDS when available) ----
#ifdef HAVE_ASYNC_LDS
    for (int i = tid; i < 63 * 16; i += 256)
        __builtin_amdgcn_global_load_async_to_lds_b32(
            (gint_as1*)(biasTab + i),
            (sint_as3*)&biasL[i], 0, 0);
#else
    for (int i = tid; i < 63 * 16; i += 256) biasL[i] = biasTab[i];
#endif

    // ---- stage 2 windows of x to LDS as bf16 (8192 float4 across 256 thr) ----
    {
        const float4* xg = (const float4*)(x + (size_t)wb * NTOK * CDIM);
        #pragma unroll
        for (int i = 0; i < 32; ++i) {
            int idx = tid + i * 256;          // float4 index, < 8192
            float4 v = xg[idx];
            int m = idx >> 7;                 // row 0..63 (window-major)
            int c = (idx << 2) & 511;
            uint2 p;
            p.x = (uint32_t)f2bf(v.x) | ((uint32_t)f2bf(v.y) << 16);
            p.y = (uint32_t)f2bf(v.z) | ((uint32_t)f2bf(v.w) << 16);
            *(uint2*)&xs[m * 520 + c] = p;
        }
    }
#ifdef HAVE_ASYNC_LDS
    asm volatile("s_wait_asynccnt 0x0" ::: "memory");
#endif
    __syncthreads();

    for (int h = 0; h < NH; ++h) {
        const float scale = scaleTab[h];

        // ---- QKV GEMM: 6 (mat,ntile) units on waves 0..5; B frag reused
        //      across 4 m-tiles (2 windows x 2 m-tiles) -> 4 accumulators ----
        if (wave < 6) {
            int sel = wave >> 1;              // 0=q 1=k 2=v
            int nt  = wave & 1;
            int rowB = sel * CDIM + h * HD + nt * 16 + l15;
            const uint4* bbase = (const uint4*)(qkvw_bf + (size_t)rowB * CDIM); // 64 uint4/row
            const uint4* abase = (const uint4*)xs;                              // 65 uint4/row
            v8f accq[4] = {};
            #pragma unroll
            for (int kk = 0; kk < 16; ++kk) {
                FragAB b;
                b.q[0] = bbase[kk * 4 + hs];
                b.q[1] = bbase[kk * 4 + hs + 2];
                #pragma unroll
                for (int mt4 = 0; mt4 < 4; ++mt4) {
                    FragAB a;
                    const uint4* ar = abase + (mt4 * 16 + l15) * 65;
                    a.q[0] = ar[kk * 4 + hs];
                    a.q[1] = ar[kk * 4 + hs + 2];
                    accq[mt4] = wmma_bf16(a, b, accq[mt4]);
                }
            }
            #pragma unroll
            for (int mt4 = 0; mt4 < 4; ++mt4) {
                if (sel == 2) {               // v: store transposed [d][m] per window
                    int g = mt4 >> 1;
                    int d = nt * 16 + l15;
                    #pragma unroll
                    for (int j = 0; j < 8; ++j) {
                        int ml = (mt4 & 1) * 16 + j + 8 * hs;
                        vh[(g * 32 + d) * 40 + ml] = f2bf(accq[mt4][j]);
                    }
                } else {                      // q/k: stage raw f32 for row norms
                    float* dst = (sel == 0) ? qf : kf;
                    int d = nt * 16 + l15;
                    #pragma unroll
                    for (int j = 0; j < 8; ++j) {
                        int m64 = mt4 * 16 + j + 8 * hs;
                        dst[m64 * 36 + d] = accq[mt4][j];
                    }
                }
            }
        }
        __syncthreads();

        // ---- L2-normalize q (fold in per-head scale) and k: 128 rows ----
        if (tid < 128) {
            int  tok = tid & 63;
            bool isq = tid < 64;
            const float* src = isq ? qf : kf;
            uint16_t*    dst = isq ? qh : kh;
            float ss = 0.0f;
            #pragma unroll
            for (int d = 0; d < HD; ++d) { float v = src[tok * 36 + d]; ss += v * v; }
            float inv = (isq ? scale : 1.0f) / fmaxf(sqrtf(ss), 1e-12f);
            #pragma unroll
            for (int d = 0; d < HD; ++d) dst[tok * 40 + d] = f2bf(src[tok * 36 + d] * inv);
        }
        __syncthreads();

        // ---- S = qn @ kn^T (+bias as C-init): 8 tiles on 8 waves ----
        {
            int g = wave >> 2, mt = (wave >> 1) & 1, nt = wave & 1;
            FragAB a, b;
            const uint4* ab = (const uint4*)qh + (g * 32 + mt * 16 + l15) * 5;
            a.q[0] = ab[hs];
            a.q[1] = ab[hs + 2];
            const uint4* bb = (const uint4*)kh + (g * 32 + nt * 16 + l15) * 5;
            b.q[0] = bb[hs];
            b.q[1] = bb[hs + 2];
            v8f c;
            int jl = nt * 16 + l15;
            #pragma unroll
            for (int j = 0; j < 8; ++j) {
                int il = mt * 16 + j + 8 * hs;
                c[j] = biasL[(il - jl + 31) * 16 + h];
            }
            c = wmma_bf16(a, b, c);
            #pragma unroll
            for (int j = 0; j < 8; ++j) {
                int r64 = g * 32 + mt * 16 + j + 8 * hs;
                sf[r64 * 36 + jl] = c[j];
            }
        }
        __syncthreads();

        // ---- softmax over 64 rows ----
        if (tid < 64) {
            float mx = -1e30f;
            #pragma unroll
            for (int j = 0; j < 32; ++j) mx = fmaxf(mx, sf[tid * 36 + j]);
            float e[32];
            float sum = 0.0f;
            #pragma unroll
            for (int j = 0; j < 32; ++j) {
                float v = __expf(sf[tid * 36 + j] - mx);
                e[j] = v; sum += v;
            }
            float inv = 1.0f / sum;
            #pragma unroll
            for (int j = 0; j < 32; ++j) sp[tid * 40 + j] = f2bf(e[j] * inv);
        }
        __syncthreads();

        // ---- O = P @ V: 8 tiles on 8 waves, store bf16 to global ao ----
        {
            int g = wave >> 2, mt = (wave >> 1) & 1, nt = wave & 1;
            FragAB a, b;
            const uint4* ab = (const uint4*)sp + (g * 32 + mt * 16 + l15) * 5;
            a.q[0] = ab[hs];
            a.q[1] = ab[hs + 2];
            const uint4* bb = (const uint4*)vh + (g * 32 + nt * 16 + l15) * 5;
            b.q[0] = bb[hs];
            b.q[1] = bb[hs + 2];
            v8f c = {};
            c = wmma_bf16(a, b, c);
            int col = h * HD + nt * 16 + l15;
            #pragma unroll
            for (int j = 0; j < 8; ++j) {
                int row = (wb + g) * NTOK + mt * 16 + j + 8 * hs;
                ao[(size_t)row * CDIM + col] = f2bf(c[j]);
            }
        }
        __syncthreads();
    }
}

// ---------------------------------------------------------------------------
// K2: out = ao(bf16) @ out_w^T + out_b, f32 output.
// Block tile 128x64; each wave owns a 32x32 quad: 4 wmma per 4 fragment
// loads (A and B each reused twice). ao (134MB) is L2-resident.
// ---------------------------------------------------------------------------
__global__ __launch_bounds__(256) void proj_kernel(
    const uint16_t* __restrict__ ao,       // [131072][512] bf16
    const uint16_t* __restrict__ outw_bf,  // [512][512] bf16 row-major
    const float*    __restrict__ out_b,    // [512]
    float*          __restrict__ out)      // [131072][512] f32
{
    const int tid  = threadIdx.x;
    const int lane = tid & 31;
    const int wave = tid >> 5;
    const int hs   = lane >> 4;
    const int l15  = lane & 15;
    const int rb = blockIdx.x >> 3, cb = blockIdx.x & 7;
    const int rowBase = rb * 128, colBase = cb * 64;
    const int mtq = wave >> 1, ctq = wave & 1;   // 32x32 quad coords

    const uint4* a0 = (const uint4*)(ao + (size_t)(rowBase + mtq * 32 + l15) * CDIM);
    const uint4* a1 = (const uint4*)(ao + (size_t)(rowBase + mtq * 32 + 16 + l15) * CDIM);
    const uint4* b0 = (const uint4*)(outw_bf + (size_t)(colBase + ctq * 32 + l15) * CDIM);
    const uint4* b1 = (const uint4*)(outw_bf + (size_t)(colBase + ctq * 32 + 16 + l15) * CDIM);

    float bv0 = out_b[colBase + ctq * 32 + l15];
    float bv1 = out_b[colBase + ctq * 32 + 16 + l15];
    v8f acc00 = { bv0, bv0, bv0, bv0, bv0, bv0, bv0, bv0 };
    v8f acc01 = { bv1, bv1, bv1, bv1, bv1, bv1, bv1, bv1 };
    v8f acc10 = acc00;
    v8f acc11 = acc01;

    #pragma unroll 4
    for (int kk = 0; kk < 16; ++kk) {
        FragAB fa0, fa1, fb0, fb1;
        fa0.q[0] = a0[kk * 4 + hs];     fa0.q[1] = a0[kk * 4 + hs + 2];
        fa1.q[0] = a1[kk * 4 + hs];     fa1.q[1] = a1[kk * 4 + hs + 2];
        fb0.q[0] = b0[kk * 4 + hs];     fb0.q[1] = b0[kk * 4 + hs + 2];
        fb1.q[0] = b1[kk * 4 + hs];     fb1.q[1] = b1[kk * 4 + hs + 2];
        acc00 = wmma_bf16(fa0, fb0, acc00);
        acc01 = wmma_bf16(fa0, fb1, acc01);
        acc10 = wmma_bf16(fa1, fb0, acc10);
        acc11 = wmma_bf16(fa1, fb1, acc11);
    }

    int c0 = colBase + ctq * 32 + l15;
    int c1 = c0 + 16;
    #pragma unroll
    for (int j = 0; j < 8; ++j) {
        int r0 = rowBase + mtq * 32 + j + 8 * hs;
        int r1 = r0 + 16;
        out[(size_t)r0 * CDIM + c0] = acc00[j];
        out[(size_t)r0 * CDIM + c1] = acc01[j];
        out[(size_t)r1 * CDIM + c0] = acc10[j];
        out[(size_t)r1 * CDIM + c1] = acc11[j];
    }
}

// ---------------------------------------------------------------------------
extern "C" void kernel_launch(void* const* d_in, const int* in_sizes, int n_in,
                              void* d_out, int out_size, void* d_ws, size_t ws_size,
                              hipStream_t stream) {
    const float* x           = (const float*)d_in[0];
    const float* qkv_w       = (const float*)d_in[1];
    const float* out_w       = (const float*)d_in[2];
    const float* out_b       = (const float*)d_in[3];
    const float* logit_scale = (const float*)d_in[4];
    const float* cpb_w1      = (const float*)d_in[5];
    const float* cpb_b1      = (const float*)d_in[6];
    const float* cpb_w2      = (const float*)d_in[7];

    char* ws = (char*)d_ws;
    float*    biasTab  = (float*)(ws + 0);                       // 63*16 f32
    float*    scaleTab = (float*)(ws + 4096);                    // 16 f32
    uint16_t* qkvw_bf  = (uint16_t*)(ws + 8192);                 // 1536*512 bf16
    uint16_t* outw_bf  = (uint16_t*)(ws + 8192 + 1536*512*2);    // 512*512 bf16
    uint16_t* ao       = (uint16_t*)(ws + 8192 + 1536*512*2 + 512*512*2); // 131072*512 bf16

    prep_convert<<<4096, 256, 0, stream>>>(qkv_w, out_w, qkvw_bf, outw_bf);
    prep_bias<<<4, 256, 0, stream>>>(cpb_w1, cpb_b1, cpb_w2, logit_scale,
                                     biasTab, scaleTab);
    attn_kernel<<<NWIN / 2, 256, 0, stream>>>(x, qkvw_bf, biasTab, scaleTab, ao);
    proj_kernel<<<(BN / 128) * (CDIM / 64), 256, 0, stream>>>(ao, outw_bf, out_b,
                                                              (float*)d_out);
}
